// QuestionPooling_5334349381886
// MI455X (gfx1250) — matrix operand order repaired and checked by reading, not compiled, not run on hardware
//
#include <hip/hip_runtime.h>
#include <hip/hip_bf16.h>
#include <math.h>

typedef __attribute__((ext_vector_type(16))) __bf16 v16bf;
typedef __attribute__((ext_vector_type(8)))  float  v8f;

#define NEG_INF_F (-1e30f)
#define B_SZ 32
#define L_SZ 4096
#define D_SZ 1024
#define A_SZ 128
#define KCHUNKS (D_SZ / 32)   // 32 K-chunks of 32
#define NTILES  (A_SZ / 16)   // 8 N-tiles of 16
#define LCHUNKS 16            // L split for the weighted-sum pass
#define LCH_SZ  (L_SZ / LCHUNKS)   // 256

// ---------------------------------------------------------------------------
// Kernel 0: pack W1 (fp32 [D=1024, A=128]) into bf16 WMMA-B layout:
//   idx = ((kc*8 + nt)*32 + lane)*16 + e
//   B element (K, N) with K = kc*32 + (lane/16)*8 + (e%8) + (e/8)*16, N = nt*16 + (lane%16)
// ---------------------------------------------------------------------------
__global__ void pack_w1_kernel(const float* __restrict__ W1,
                               unsigned short* __restrict__ W1p) {
    int tid = blockIdx.x * 256 + threadIdx.x;      // 0 .. 131071
    int e    = tid & 15;
    int lane = (tid >> 4) & 31;
    int nt   = (tid >> 9) & 7;
    int kc   = tid >> 12;
    int k = kc * 32 + (lane >> 4) * 8 + (e & 7) + (e >> 3) * 16;
    int n = nt * 16 + (lane & 15);
    union { __bf16 h; unsigned short u; } cv;
    cv.h = (__bf16)W1[k * A_SZ + n];
    W1p[tid] = cv.u;
}

// ---------------------------------------------------------------------------
// A-operand builder: 16-bit A layout, lane (0-15) = row M, khalf = lane/16.
// Elements 0..7  -> K = kbase + 0..7
// Elements 8..15 -> K = kbase + 16..23      (kbase = kc*32 + 8*khalf)
// ---------------------------------------------------------------------------
__device__ __forceinline__ v16bf load_a_tile(const float* __restrict__ pr) {
    float4 f0 = *(const float4*)(pr);
    float4 f1 = *(const float4*)(pr + 4);
    float4 f2 = *(const float4*)(pr + 16);
    float4 f3 = *(const float4*)(pr + 20);
    v16bf a;
    a[0]  = (__bf16)f0.x; a[1]  = (__bf16)f0.y; a[2]  = (__bf16)f0.z; a[3]  = (__bf16)f0.w;
    a[4]  = (__bf16)f1.x; a[5]  = (__bf16)f1.y; a[6]  = (__bf16)f1.z; a[7]  = (__bf16)f1.w;
    a[8]  = (__bf16)f2.x; a[9]  = (__bf16)f2.y; a[10] = (__bf16)f2.z; a[11] = (__bf16)f2.w;
    a[12] = (__bf16)f3.x; a[13] = (__bf16)f3.y; a[14] = (__bf16)f3.z; a[15] = (__bf16)f3.w;
    return a;
}

// ---------------------------------------------------------------------------
// Kernel 1: fused GEMM + tanh + w2 reduction -> s[B*L]
// Each wave: 32 rows (two 16-row M-tiles) x full A=128 (8 N-tiles),
// K accumulated over 32 chunks of 32 with v_wmma_f32_16x16x32_bf16.
// 8 waves share B chunks via DOUBLE-BUFFERED LDS: one barrier per K-chunk;
// next chunk's stores overlap current chunk's ds_load + wmma.
// ---------------------------------------------------------------------------
__global__ void __launch_bounds__(256)
gemm_s_kernel(const float* __restrict__ u_q,
              const unsigned short* __restrict__ W1p,
              const float* __restrict__ b1,
              const float* __restrict__ w2,
              const float* __restrict__ b2,
              float* __restrict__ s_out) {
    __shared__ __align__(32) __bf16 Bsh[2][NTILES * 512];   // 2 x 8 KB

    const int wave = threadIdx.x >> 5;
    const int lane = threadIdx.x & 31;
    const int khalf = lane >> 4;
    const int row_base = (blockIdx.x * 8 + wave) * 32;

    const size_t row0 = (size_t)(row_base + (lane & 15));
    const size_t row1 = row0 + 16;
    const float* pr0_base = u_q + row0 * D_SZ + khalf * 8;
    const float* pr1_base = u_q + row1 * D_SZ + khalf * 8;

    const v8f vzero = {0.f, 0.f, 0.f, 0.f, 0.f, 0.f, 0.f, 0.f};
    v8f acc0[NTILES], acc1[NTILES];
#pragma unroll
    for (int nt = 0; nt < NTILES; ++nt) { acc0[nt] = vzero; acc1[nt] = vzero; }

    // Pre-stage chunk 0 into buffer 0 (32 B per thread, contiguous packed layout)
    *(v16bf*)&Bsh[0][threadIdx.x * 16] =
        *(const v16bf*)(W1p + threadIdx.x * 16);

    for (int kc = 0; kc < KCHUNKS; ++kc) {
        // One barrier per chunk:
        //  - publishes stores into Bsh[kc&1] issued last iteration
        //  - guarantees all waves finished reading Bsh[(kc+1)&1] (read in kc-1)
        __syncthreads();
        if (kc + 1 < KCHUNKS) {
            *(v16bf*)&Bsh[(kc + 1) & 1][threadIdx.x * 16] =
                *(const v16bf*)(W1p + (size_t)(kc + 1) * 4096 + threadIdx.x * 16);
        }

        const float* pr0 = pr0_base + kc * 32;
        const float* pr1 = pr1_base + kc * 32;
        if (kc + 1 < KCHUNKS) {                  // hint next A chunk
            __builtin_prefetch(pr0 + 32, 0, 1);
            __builtin_prefetch(pr1 + 32, 0, 1);
        }
        v16bf a0 = load_a_tile(pr0);
        v16bf a1 = load_a_tile(pr1);

        const __bf16* bcur = &Bsh[kc & 1][lane * 16];
#pragma unroll
        for (int nt = 0; nt < NTILES; ++nt) {
            v16bf bt = *(const v16bf*)(bcur + nt * 512);
            acc0[nt] = __builtin_amdgcn_wmma_f32_16x16x32_bf16(
                false, a0, false, bt, (short)0, acc0[nt], false, false);
            acc1[nt] = __builtin_amdgcn_wmma_f32_16x16x32_bf16(
                false, a1, false, bt, (short)0, acc1[nt], false, false);
        }
    }

    // Epilogue: C layout -> VGPR j holds row (j + 8*khalf), col = nt*16 + (lane&15)
    const int nl = lane & 15;
    float s0[8], s1[8];
#pragma unroll
    for (int j = 0; j < 8; ++j) { s0[j] = 0.f; s1[j] = 0.f; }
#pragma unroll
    for (int nt = 0; nt < NTILES; ++nt) {
        const int ai = nt * 16 + nl;
        const float wv = w2[ai];
        const float bv = b1[ai];
#pragma unroll
        for (int j = 0; j < 8; ++j) {
            s0[j] += wv * tanhf(acc0[nt][j] + bv);
            s1[j] += wv * tanhf(acc1[nt][j] + bv);
        }
    }
    // reduce across 16 lanes within each half (offsets < 16 stay in-half)
#pragma unroll
    for (int j = 0; j < 8; ++j) {
#pragma unroll
        for (int off = 1; off < 16; off <<= 1) {
            s0[j] += __shfl_xor(s0[j], off, 32);
            s1[j] += __shfl_xor(s1[j], off, 32);
        }
    }
    const float bias2 = b2[0];
    if (lane == 0) {
#pragma unroll
        for (int j = 0; j < 8; ++j) {
            s_out[row_base + j]      = s0[j] + bias2;   // tile0 rows 0..7
            s_out[row_base + 16 + j] = s1[j] + bias2;   // tile1 rows 0..7
        }
    } else if (lane == 16) {
#pragma unroll
        for (int j = 0; j < 8; ++j) {
            s_out[row_base + 8 + j]  = s0[j] + bias2;   // tile0 rows 8..15
            s_out[row_base + 24 + j] = s1[j] + bias2;   // tile1 rows 8..15
        }
    }
}

// ---------------------------------------------------------------------------
// Kernel 2: masked softmax over L per batch row
// ---------------------------------------------------------------------------
__global__ void softmax_kernel(const float* __restrict__ s,
                               const unsigned char* __restrict__ mask,
                               float* __restrict__ a) {
    __shared__ float red[256];
    const int b = blockIdx.x;
    const int t = threadIdx.x;
    const size_t base = (size_t)b * L_SZ;

    float m = -3.0e38f;
    for (int l = t; l < L_SZ; l += 256) {
        float v = mask[base + l] ? s[base + l] : NEG_INF_F;
        m = fmaxf(m, v);
    }
    red[t] = m;
    __syncthreads();
    for (int o = 128; o > 0; o >>= 1) {
        if (t < o) red[t] = fmaxf(red[t], red[t + o]);
        __syncthreads();
    }
    m = red[0];
    __syncthreads();

    float sum = 0.f;
    for (int l = t; l < L_SZ; l += 256) {
        float v = mask[base + l] ? s[base + l] : NEG_INF_F;
        float e = expf(v - m);
        a[base + l] = e;
        sum += e;
    }
    red[t] = sum;
    __syncthreads();
    for (int o = 128; o > 0; o >>= 1) {
        if (t < o) red[t] += red[t + o];
        __syncthreads();
    }
    const float inv = 1.0f / red[0];
    for (int l = t; l < L_SZ; l += 256) a[base + l] *= inv;
}

// ---------------------------------------------------------------------------
// Kernel 3: partial weighted sums  part[b][chunk][d] = sum_{l in chunk} a*u_q
// grid = 32*16, block = 256 threads, thread owns d = 4t..4t+3 (float4 stream)
// ---------------------------------------------------------------------------
__global__ void wsum_partial_kernel(const float* __restrict__ u_q,
                                    const float* __restrict__ a,
                                    float* __restrict__ part) {
    __shared__ float ash[LCH_SZ];
    const int b = blockIdx.x >> 4;
    const int chunk = blockIdx.x & 15;
    const int t = threadIdx.x;
    const int l0 = chunk * LCH_SZ;

    if (t < LCH_SZ) ash[t] = a[(size_t)b * L_SZ + l0 + t];
    __syncthreads();

    float4 acc = {0.f, 0.f, 0.f, 0.f};
    const float* base = u_q + ((size_t)b * L_SZ + l0) * D_SZ + 4 * t;
    for (int l = 0; l < LCH_SZ; ++l) {
        const float av = ash[l];
        float4 q = *(const float4*)(base + (size_t)l * D_SZ);
        acc.x += av * q.x; acc.y += av * q.y;
        acc.z += av * q.z; acc.w += av * q.w;
    }
    *(float4*)(part + ((size_t)(b * LCHUNKS + chunk)) * D_SZ + 4 * t) = acc;
}

// ---------------------------------------------------------------------------
// Kernel 4: reduce partials -> out[B*D]
// ---------------------------------------------------------------------------
__global__ void reduce_out_kernel(const float* __restrict__ part,
                                  float* __restrict__ out) {
    const int idx = blockIdx.x * 256 + threadIdx.x;   // 0..32767
    const int b = idx >> 10;
    const int d = idx & 1023;
    float s = 0.f;
#pragma unroll
    for (int c = 0; c < LCHUNKS; ++c)
        s += part[((size_t)(b * LCHUNKS + c) << 10) + d];
    out[idx] = s;
}

// ---------------------------------------------------------------------------
extern "C" void kernel_launch(void* const* d_in, const int* in_sizes, int n_in,
                              void* d_out, int out_size, void* d_ws, size_t ws_size,
                              hipStream_t stream) {
    const float*         u_q  = (const float*)d_in[0];
    const unsigned char* mask = (const unsigned char*)d_in[1];
    const float*         W1   = (const float*)d_in[2];
    const float*         b1   = (const float*)d_in[3];
    const float*         w2   = (const float*)d_in[4];
    const float*         b2   = (const float*)d_in[5];
    float* out = (float*)d_out;

    char* ws = (char*)d_ws;
    unsigned short* W1p = (unsigned short*)ws;            // 262144 B
    float* s    = (float*)(ws + 262144);                  // 524288 B
    float* a    = (float*)(ws + 786432);                  // 524288 B
    float* part = (float*)(ws + 1310720);                 // 2097152 B (32*16*1024 f32)

    pack_w1_kernel<<<512, 256, 0, stream>>>(W1, W1p);
    gemm_s_kernel<<<512, 256, 0, stream>>>(u_q, W1p, b1, w2, b2, s);
    softmax_kernel<<<B_SZ, 256, 0, stream>>>(s, mask, a);
    wsum_partial_kernel<<<B_SZ * LCHUNKS, 256, 0, stream>>>(u_q, a, part);
    reduce_out_kernel<<<128, 256, 0, stream>>>(part, out);
}
